// ContextualAttention_49022756716702
// MI455X (gfx1250) — matrix operand order repaired and checked by reading, not compiled, not run on hardware
//
#include <hip/hip_runtime.h>
#include <hip/hip_bf16.h>
#include <math.h>

typedef __attribute__((ext_vector_type(16))) _Float16 v16h;
typedef __attribute__((ext_vector_type(8)))  _Float16 v8h;
typedef __attribute__((ext_vector_type(8)))  float    v8f;

#define HD     64
#define DIMF   384
#define KSPLIT 4          // waves per 16-query tile (split-K flash)

// fold 1/sqrt(HD) and log2(e) into Q so the softmax runs in base 2
// (uniform base change cancels in softmax): s2 = (q.k/8)*log2(e)
#define QSCALE (0.125f * 1.4426950408889634f)

union V16U { v16h v; v8h g[2]; };

// ---------------------------------------------------------------------------
// Kernel 1: turn_hidden = [emb; bil] @ Wt^T + bt  (fp32),
//           q = (th @ Wq^T + bq) * QSCALE -> f16 rowmajor [N][64]
//           k = (th @ Wk^T + bk)          -> f16 rowmajor [N][64]
//           v = (th @ Wv^T + bv)          -> f16 TRANSPOSED [64][N]
// ---------------------------------------------------------------------------
__global__ __launch_bounds__(256)
void ca_proj_kernel(const float* __restrict__ emb,
                    const float* __restrict__ bil,
                    const float* __restrict__ Wt, const float* __restrict__ bt,
                    const float* __restrict__ Wq, const float* __restrict__ bq,
                    const float* __restrict__ Wk, const float* __restrict__ bk,
                    const float* __restrict__ Wv, const float* __restrict__ bv,
                    float* __restrict__ th,
                    _Float16* __restrict__ q16,
                    _Float16* __restrict__ k16,
                    _Float16* __restrict__ vT16,
                    int N) {
  __shared__ float s_row[4][DIMF];
  __shared__ float s_th[4][HD];
  const int tid  = threadIdx.x;
  const int r    = tid >> 6;
  const int t    = tid & 63;
  const int row0 = blockIdx.x * 4;

  for (int i = tid; i < 4 * DIMF; i += 256)
    s_row[i / DIMF][i % DIMF] = emb[(size_t)row0 * DIMF + i];
  __syncthreads();

  const int row = row0 + r;
  const float* wrow = Wt + (size_t)t * (DIMF + 1);
  float acc = bt[t];
  #pragma unroll 8
  for (int j = 0; j < DIMF; ++j) acc += s_row[r][j] * wrow[j];
  acc += bil[row] * wrow[DIMF];
  s_th[r][t] = acc;
  th[(size_t)row * HD + t] = acc;
  __syncthreads();

  float aq = bq[t], ak = bk[t], av = bv[t];
  #pragma unroll 8
  for (int j = 0; j < HD; ++j) {
    float hv = s_th[r][j];
    aq += hv * Wq[t * HD + j];
    ak += hv * Wk[t * HD + j];
    av += hv * Wv[t * HD + j];
  }
  q16[(size_t)row * HD + t] = (_Float16)(aq * QSCALE);
  k16[(size_t)row * HD + t] = (_Float16)ak;
  vT16[(size_t)t * N + row] = (_Float16)av;
}

// ---------------------------------------------------------------------------
// Kernel 2: situation path, algebraically collapsed:
//   sh = Ws@sit + bs; ck = Wck@sh + bck; cv = Wcv@sh + bcv
//   u[j] = sum_t Wcq[t][j]*ck[t];  c0 = bcq.ck;  vw = cv.Wsc
// ---------------------------------------------------------------------------
__global__ __launch_bounds__(64)
void ca_sit_kernel(const float* __restrict__ sit,
                   const float* __restrict__ Ws,  const float* __restrict__ bs,
                   const float* __restrict__ Wck, const float* __restrict__ bck,
                   const float* __restrict__ Wcv, const float* __restrict__ bcv,
                   const float* __restrict__ Wcq, const float* __restrict__ bcq,
                   const float* __restrict__ Wsc,
                   float* __restrict__ u, float* __restrict__ scal) {
  __shared__ float sh[HD], ck[HD], cv[HD];
  const int t = threadIdx.x;
  float a = bs[t];
  for (int j = 0; j < DIMF; ++j) a += sit[j] * Ws[t * DIMF + j];
  sh[t] = a;
  __syncthreads();
  float kk = bck[t], vv = bcv[t];
  for (int j = 0; j < HD; ++j) {
    kk += sh[j] * Wck[t * HD + j];
    vv += sh[j] * Wcv[t * HD + j];
  }
  ck[t] = kk; cv[t] = vv;
  __syncthreads();
  float uu = 0.0f;
  for (int tt = 0; tt < HD; ++tt) uu += Wcq[tt * HD + t] * ck[tt];
  u[t] = uu;
  if (t == 0) {
    float c0 = 0.0f, vw = 0.0f;
    for (int tt = 0; tt < HD; ++tt) { c0 += bcq[tt] * ck[tt]; vw += cv[tt] * Wsc[tt]; }
    scal[0] = c0; scal[1] = vw;
  }
}

// ---------------------------------------------------------------------------
// Kernel 3: split-K flash attention. Block = 4 waves, all serving the same
// 16-query tile; wave w handles keys [w*N/4, (w+1)*N/4). Partials (m, l, O)
// merged in LDS with the log-sum-exp combine. S^T = K*Q^T trick keeps the
// wmma accumulator registers in exactly the A-operand layout for P@V.
// Base-2 softmax (scale folded into Q). Lazy max-rescale behind __any().
// th updated in place: th += softmax(QK^T/s) @ V.
// ---------------------------------------------------------------------------
__global__ __launch_bounds__(32 * KSPLIT)
void ca_flash_kernel(const _Float16* __restrict__ q16,
                     const _Float16* __restrict__ k16,
                     const _Float16* __restrict__ vT16,
                     float* __restrict__ th,
                     int N) {
  __shared__ float sO[KSPLIT][16][HD];
  __shared__ float sm[KSPLIT][16];
  __shared__ float sl[KSPLIT][16];

  const int tid   = threadIdx.x;
  const int wave  = tid >> 5;
  const int lane  = tid & 31;
  const int n     = lane & 15;        // B column / C lane index
  const int h     = lane >> 4;        // lane half
  const int h8    = h << 3;
  const int qbase = blockIdx.x * 16;

  // Q as B-operand: lane n holds Q[qbase+n][k0..k15] per chunk (32B each).
  const v16h* qrow = (const v16h*)(q16 + (size_t)(qbase + n) * HD);
  const v16h qb0 = qrow[h];       // head dims  0..31 chunk
  const v16h qb1 = qrow[2 + h];   // head dims 32..63 chunk

  v8f O0 = {}, O1 = {}, O2 = {}, O3 = {};
  float m_run  = -1e30f;
  float l_lane = 0.0f;

  const int kper = N / KSPLIT;
  const int kbeg = wave * kper;
  const int kend = kbeg + kper;

  for (int j = kbeg; j < kend; j += 32) {
    // K tiles as A-operand: lane m(=n) holds 8-elem groups {2h,2h+1}/{4+2h,5+2h}.
    const v8h* kr0 = (const v8h*)(k16 + (size_t)(j + n) * HD);
    const v8h* kr1 = (const v8h*)(k16 + (size_t)(j + 16 + n) * HD);
    V16U ka0, ka1, kb0, kb1;
    ka0.g[0] = kr0[2 * h];     ka0.g[1] = kr0[2 * h + 1];
    ka1.g[0] = kr0[4 + 2 * h]; ka1.g[1] = kr0[5 + 2 * h];
    kb0.g[0] = kr1[2 * h];     kb0.g[1] = kr1[2 * h + 1];
    kb1.g[0] = kr1[4 + 2 * h]; kb1.g[1] = kr1[5 + 2 * h];
    if (j + 32 < kend) __builtin_prefetch(k16 + (size_t)(j + 32 + n) * HD, 0, 1);

    v8f zc = {};
    v8f S0 = __builtin_amdgcn_wmma_f32_16x16x32_f16(false, ka0.v, false, qb0, (short)0, zc, false, false);
    S0     = __builtin_amdgcn_wmma_f32_16x16x32_f16(false, ka1.v, false, qb1, (short)0, S0, false, false);
    v8f S1 = __builtin_amdgcn_wmma_f32_16x16x32_f16(false, kb0.v, false, qb0, (short)0, zc, false, false);
    S1     = __builtin_amdgcn_wmma_f32_16x16x32_f16(false, kb1.v, false, qb1, (short)0, S1, false, false);

    // chunk max for this lane's 16 scores, combined across the lane pair
    float cmax = fmaxf(S0[0], S1[0]);
    #pragma unroll
    for (int e = 1; e < 8; ++e) cmax = fmaxf(cmax, fmaxf(S0[e], S1[e]));
    cmax = fmaxf(cmax, __shfl_xor(cmax, 16, 32));

    // lazy rescale: only when some query's max actually grows (rare)
    if (__any(cmax > m_run)) {
      const float mnew  = fmaxf(m_run, cmax);
      const float alpha = __builtin_amdgcn_exp2f(m_run - mnew);
      m_run  = mnew;
      l_lane *= alpha;
      float arow[8];
      #pragma unroll
      for (int rr = 0; rr < 8; ++rr) arow[rr] = __shfl(alpha, rr + h8, 32);
      #pragma unroll
      for (int rr = 0; rr < 8; ++rr) {
        O0[rr] *= arow[rr]; O1[rr] *= arow[rr];
        O2[rr] *= arow[rr]; O3[rr] *= arow[rr];
      }
    }

    v16h P;                       // already in A-operand layout for P@V
    float lsum = 0.0f;
    #pragma unroll
    for (int e = 0; e < 8; ++e) {
      float p0 = __builtin_amdgcn_exp2f(S0[e] - m_run);
      float p1 = __builtin_amdgcn_exp2f(S1[e] - m_run);
      lsum += p0 + p1;
      P[e]     = (_Float16)p0;
      P[8 + e] = (_Float16)p1;
    }
    l_lane += lsum;

    // V as B-operand: two 16B loads per tile, keys 8h..8h+7 and 16+8h..16+8h+7.
    const size_t vcol = (size_t)j + (size_t)h * 8;
    const v16h Vb0 = __builtin_shufflevector(*(const v8h*)(vT16 + (size_t)(0 * 16 + n) * N + vcol), *(const v8h*)(vT16 + (size_t)(0 * 16 + n) * N + vcol + 16), 0,1,2,3,4,5,6,7,8,9,10,11,12,13,14,15);
    const v16h Vb1 = __builtin_shufflevector(*(const v8h*)(vT16 + (size_t)(1 * 16 + n) * N + vcol), *(const v8h*)(vT16 + (size_t)(1 * 16 + n) * N + vcol + 16), 0,1,2,3,4,5,6,7,8,9,10,11,12,13,14,15);
    const v16h Vb2 = __builtin_shufflevector(*(const v8h*)(vT16 + (size_t)(2 * 16 + n) * N + vcol), *(const v8h*)(vT16 + (size_t)(2 * 16 + n) * N + vcol + 16), 0,1,2,3,4,5,6,7,8,9,10,11,12,13,14,15);
    const v16h Vb3 = __builtin_shufflevector(*(const v8h*)(vT16 + (size_t)(3 * 16 + n) * N + vcol), *(const v8h*)(vT16 + (size_t)(3 * 16 + n) * N + vcol + 16), 0,1,2,3,4,5,6,7,8,9,10,11,12,13,14,15);
    O0 = __builtin_amdgcn_wmma_f32_16x16x32_f16(false, P, false, Vb0, (short)0, O0, false, false);
    O1 = __builtin_amdgcn_wmma_f32_16x16x32_f16(false, P, false, Vb1, (short)0, O1, false, false);
    O2 = __builtin_amdgcn_wmma_f32_16x16x32_f16(false, P, false, Vb2, (short)0, O2, false, false);
    O3 = __builtin_amdgcn_wmma_f32_16x16x32_f16(false, P, false, Vb3, (short)0, O3, false, false);
  }

  // publish this wave's partial (m, l, O) to LDS
  const float l_tot = l_lane + __shfl_xor(l_lane, 16, 32);
  if (h == 0) { sm[wave][n] = m_run; sl[wave][n] = l_tot; }
  #pragma unroll
  for (int rr = 0; rr < 8; ++rr) {
    const int q = rr + h8;                       // C row M = rr + 8*h
    sO[wave][q][0 * 16 + n] = O0[rr];
    sO[wave][q][1 * 16 + n] = O1[rr];
    sO[wave][q][2 * 16 + n] = O2[rr];
    sO[wave][q][3 * 16 + n] = O3[rr];
  }
  __syncthreads();

  // merge KSPLIT partials: O* = sum_w O_w * 2^(m_w-m*),  l* likewise
  for (int e = tid; e < 16 * HD; e += 32 * KSPLIT) {
    const int q  = e >> 6;
    const int hd = e & (HD - 1);
    float m0 = sm[0][q], m1 = sm[1][q], m2 = sm[2][q], m3 = sm[3][q];
    const float ms = fmaxf(fmaxf(m0, m1), fmaxf(m2, m3));
    const float w0 = __builtin_amdgcn_exp2f(m0 - ms);
    const float w1 = __builtin_amdgcn_exp2f(m1 - ms);
    const float w2 = __builtin_amdgcn_exp2f(m2 - ms);
    const float w3 = __builtin_amdgcn_exp2f(m3 - ms);
    const float lt = sl[0][q] * w0 + sl[1][q] * w1 + sl[2][q] * w2 + sl[3][q] * w3;
    const float ov = sO[0][q][hd] * w0 + sO[1][q][hd] * w1 +
                     sO[2][q][hd] * w2 + sO[3][q][hd] * w3;
    float* dst = th + (size_t)(qbase + q) * HD + hd;
    *dst = fmaf(ov, __builtin_amdgcn_rcpf(lt), *dst);
  }
}

// ---------------------------------------------------------------------------
// Kernel 4: per-row tail. score = th2.w + sigmoid((th2.u + c0)/8)*vw + bsc
//           out = (1-g)*bil + g*score,  g = sigmoid(residual_gate)
// ---------------------------------------------------------------------------
__global__ __launch_bounds__(256)
void ca_final_kernel(const float* __restrict__ th2,
                     const float* __restrict__ u,
                     const float* __restrict__ scal,
                     const float* __restrict__ Wsc,
                     const float* __restrict__ bsc,
                     const float* __restrict__ bil,
                     const float* __restrict__ rgate,
                     float* __restrict__ out,
                     int N) {
  __shared__ float su[HD], sw[HD];
  if (threadIdx.x < HD) { su[threadIdx.x] = u[threadIdx.x]; sw[threadIdx.x] = Wsc[threadIdx.x]; }
  __syncthreads();
  const int i = blockIdx.x * 256 + threadIdx.x;
  if (i >= N) return;
  const float* row = th2 + (size_t)i * HD;
  float d1 = 0.0f, d2 = 0.0f;
  #pragma unroll 8
  for (int j = 0; j < HD; ++j) { float x = row[j]; d1 += x * su[j]; d2 += x * sw[j]; }
  const float cw    = 1.0f / (1.0f + __expf(-(d1 + scal[0]) * 0.125f));
  const float score = d2 + cw * scal[1] + bsc[0];
  const float g     = 1.0f / (1.0f + __expf(-rgate[0]));
  out[i] = (1.0f - g) * bil[i] + g * score;
}

extern "C" void kernel_launch(void* const* d_in, const int* in_sizes, int n_in,
                              void* d_out, int out_size, void* d_ws, size_t ws_size,
                              hipStream_t stream) {
  const float* situation = (const float*)d_in[0];
  const float* emb       = (const float*)d_in[1];
  const float* bil       = (const float*)d_in[2];
  const float* Wt   = (const float*)d_in[3];
  const float* bt   = (const float*)d_in[4];
  const float* Ws   = (const float*)d_in[5];
  const float* bs   = (const float*)d_in[6];
  const float* Wsaq = (const float*)d_in[7];
  const float* bsaq = (const float*)d_in[8];
  const float* Wsak = (const float*)d_in[9];
  const float* bsak = (const float*)d_in[10];
  const float* Wsav = (const float*)d_in[11];
  const float* bsav = (const float*)d_in[12];
  const float* Wcq  = (const float*)d_in[13];
  const float* bcq  = (const float*)d_in[14];
  const float* Wck  = (const float*)d_in[15];
  const float* bck  = (const float*)d_in[16];
  const float* Wcv  = (const float*)d_in[17];
  const float* bcv  = (const float*)d_in[18];
  const float* Wsc  = (const float*)d_in[19];
  const float* bsc  = (const float*)d_in[20];
  const float* rg   = (const float*)d_in[21];
  float* out = (float*)d_out;

  const int N = in_sizes[2];   // 8192

  // Workspace carve-up (all offsets well aligned).
  char* ws = (char*)d_ws;
  float*     th  = (float*)ws;                                   // N*64 f32
  _Float16*  q16 = (_Float16*)(ws + (size_t)N * HD * 4);         // N*64 f16
  _Float16*  k16 = (_Float16*)(ws + (size_t)N * HD * 6);         // N*64 f16
  _Float16*  vT  = (_Float16*)(ws + (size_t)N * HD * 8);         // 64*N f16
  float*     u   = (float*)(ws + (size_t)N * HD * 10);           // 64 f32
  float*     scl = u + HD;                                       // 2  f32

  ca_proj_kernel<<<N / 4, 256, 0, stream>>>(emb, bil, Wt, bt,
                                            Wsaq, bsaq, Wsak, bsak, Wsav, bsav,
                                            th, q16, k16, vT, N);
  ca_sit_kernel<<<1, 64, 0, stream>>>(situation, Ws, bs, Wck, bck, Wcv, bcv,
                                      Wcq, bcq, Wsc, u, scl);
  ca_flash_kernel<<<N / 16, 32 * KSPLIT, 0, stream>>>(q16, k16, vT, th, N);
  ca_final_kernel<<<(N + 255) / 256, 256, 0, stream>>>(th, u, scl, Wsc, bsc,
                                                       bil, rg, out, N);
}
